// Attention_61117384622413
// MI455X (gfx1250) — matrix operand order repaired and checked by reading, not compiled, not run on hardware
//
#include <hip/hip_runtime.h>
#include <stdint.h>

// ---------------------------------------------------------------------------
// Bahdanau attention, fused for MI455X (gfx1250, wave32, WMMA).
//   B=32, S=1024, H=1024
//   scores GEMM in bf16 WMMA (v_wmma_f32_16x16x32_bf16), register-blocked
//   4(M) x 4(N) per wave so each k-step does 16 WMMAs on 8 fragment loads,
//   fused with tanh + v-dot so W_s [B,S,H] (128 MB) never touches HBM.
// ---------------------------------------------------------------------------

typedef __attribute__((ext_vector_type(16))) __bf16 v16bf;
typedef __attribute__((ext_vector_type(8)))  float  v8f;

union Frag {
    uint32_t u[8];
    uint4    q[2];
    v16bf    v;
};

#define H_DIM   1024
#define S_DIM   1024
#define B_DIM   32
#define SROWS   64          // S rows per block in the scores kernel
#define PADU    516         // uints per LDS row (1032 bf16; pad breaks bank conflicts)

__device__ __forceinline__ uint32_t pack_bf16x2(float a, float b) {
    uint32_t ua = __float_as_uint(a);
    uint32_t ub = __float_as_uint(b);
    ua += 0x7FFFu + ((ua >> 16) & 1u);   // round-to-nearest-even
    ub += 0x7FFFu + ((ub >> 16) & 1u);
    return (ua >> 16) | (ub & 0xFFFF0000u);
}

// Branch-free tanh: 1 - 2/(exp(2x)+1). exp->inf => 1, exp->0 => -1.
__device__ __forceinline__ float tanh_fast(float x) {
    float e = __expf(2.0f * x);
    return 1.0f - 2.0f * __builtin_amdgcn_rcpf(e + 1.0f);
}

// ---- Kernel 1: convert W_w [H,H] fp32 -> packed bf16x2 rows --------------
__global__ __launch_bounds__(256) void convw_kernel(const float* __restrict__ W,
                                                    uint32_t* __restrict__ Wbf) {
    int i = blockIdx.x * 256 + threadIdx.x;        // over H*H/4 float4s
    float4 f = ((const float4*)W)[i];
    Wbf[i * 2 + 0] = pack_bf16x2(f.x, f.y);
    Wbf[i * 2 + 1] = pack_bf16x2(f.z, f.w);
}

// ---- Kernel 2: U_h[b,o] = sum_h dh[b,h] * U_w[o,h] -----------------------
__global__ __launch_bounds__(256) void uh_kernel(const float* __restrict__ dh,
                                                 const float* __restrict__ Uw,
                                                 float* __restrict__ Uh) {
    __shared__ float sdh[H_DIM];
    const int b = blockIdx.y;
    const int o = blockIdx.x * 256 + threadIdx.x;
    for (int i = threadIdx.x; i < H_DIM; i += 256) sdh[i] = dh[b * H_DIM + i];
    __syncthreads();
    const float* row = Uw + (size_t)o * H_DIM;
    float acc = 0.f;
    #pragma unroll 8
    for (int h = 0; h < H_DIM; ++h) acc = fmaf(sdh[h], row[h], acc);
    Uh[b * H_DIM + o] = acc;
}

// ---- Kernel 3: fused scores: att[b,s] = v . tanh(U_h[b] + enc[b,s] @ W^T) -
// grid (S/SROWS, B), 512 threads = 16 waves. Dynamic LDS:
//   [0, SROWS*PADU) uints : bf16 A tile (padded row stride)
//   then SROWS floats     : per-row score accumulator
__global__ __launch_bounds__(512) void attn_scores_kernel(
        const float*    __restrict__ enc,   // [B,S,H] fp32
        const uint32_t* __restrict__ Wbf,   // [H, H/2] packed bf16x2 (rows of W_w)
        const float*    __restrict__ Uh,    // [B,H]
        const float*    __restrict__ vw,    // [H]
        float*          __restrict__ att) { // [B,S]
    extern __shared__ uint32_t smem[];
    uint32_t* ldsA   = smem;                         // SROWS * PADU uints
    float*    ldsAtt = (float*)(smem + SROWS * PADU);

    const int b    = blockIdx.y;
    const int s0   = blockIdx.x * SROWS;
    const int tid  = threadIdx.x;
    const int lane = tid & 31;
    const int wave = tid >> 5;          // 0..15
    const int l16  = lane & 15;
    const int hsel = lane >> 4;         // 0: low K half, 1: high K half

    // Stage 64x1024 enc tile, converting fp32 -> bf16 into padded LDS.
    const float4* src = (const float4*)(enc + ((size_t)b * S_DIM + s0) * H_DIM);
    for (int i = tid; i < SROWS * (H_DIM / 4); i += 512) {
        int row = i >> 8;               // H/4 = 256 float4 per row
        int c4  = i & 255;
        float4 f = src[row * 256 + c4];
        ldsA[row * PADU + c4 * 2 + 0] = pack_bf16x2(f.x, f.y);
        ldsA[row * PADU + c4 * 2 + 1] = pack_bf16x2(f.z, f.w);
    }
    if (tid < SROWS) ldsAtt[tid] = 0.0f;
    __syncthreads();

    // This wave's 4 output columns (one per N-tile): n_tile = wave + 16*j.
    int   ncol[4];
    float uh[4], vv[4];
    #pragma unroll
    for (int j = 0; j < 4; ++j) {
        ncol[j] = (wave + 16 * j) * 16 + l16;
        uh[j]   = Uh[b * H_DIM + ncol[j]];
        vv[j]   = vw[ncol[j]];
    }

    v8f acc[4][4];                      // [m_sub][n_tile]
    #pragma unroll
    for (int m = 0; m < 4; ++m)
        #pragma unroll
        for (int j = 0; j < 4; ++j) acc[m][j] = (v8f){};

    for (int ks = 0; ks < 32; ++ks) {   // K = 1024 in steps of 32
        // B frags: lane holds W_w row n, K-chunk [k0 + hsel*16, +16) (contiguous k order, same as A)
        Frag bf[4];
        #pragma unroll
        for (int j = 0; j < 4; ++j) {
            const uint32_t* wrow =
                Wbf + (size_t)ncol[j] * (H_DIM / 2) + hsel * 8 + ks * 16;
            bf[j].q[0] = *(const uint4*)(wrow);
            bf[j].q[1] = *(const uint4*)(wrow + 4);
        }
        #pragma unroll
        for (int m = 0; m < 4; ++m) {
            // A frag: lane row = m*16 + l16; K chunk [k0 + hsel*16, +16), matching B:
            // lanes 0-15: K 0-15 ; lanes 16-31: K 16-31
            const uint32_t* arow = ldsA + (m * 16 + l16) * PADU + ks * 16 + hsel * 8;
            Frag af;
            af.q[0] = *(const uint4*)(arow);
            af.q[1] = *(const uint4*)(arow + 4);
            #pragma unroll
            for (int j = 0; j < 4; ++j)
                acc[m][j] = __builtin_amdgcn_wmma_f32_16x16x32_bf16(
                                false, af.v, false, bf[j].v, (short)0, acc[m][j],
                                false, false);
        }
    }

    // Epilogue: C layout — lane = N (mod 16), VGPR r = row m (+8 for upper lanes).
    float rowsum[4][8];
    #pragma unroll
    for (int m = 0; m < 4; ++m)
        #pragma unroll
        for (int r = 0; r < 8; ++r) {
            float s = 0.f;
            #pragma unroll
            for (int j = 0; j < 4; ++j)
                s += tanh_fast(acc[m][j][r] + uh[j]) * vv[j];
            rowsum[m][r] = s;
        }

    // Reduce the 16 N-lanes of each group, then combine waves via LDS atomics.
    #pragma unroll
    for (int m = 0; m < 4; ++m)
        #pragma unroll
        for (int r = 0; r < 8; ++r) {
            float v = rowsum[m][r];
            #pragma unroll
            for (int off = 8; off; off >>= 1) v += __shfl_xor(v, off, 16);
            rowsum[m][r] = v;
        }
    if (l16 == 0) {
        #pragma unroll
        for (int m = 0; m < 4; ++m)
            #pragma unroll
            for (int r = 0; r < 8; ++r)
                atomicAdd(&ldsAtt[m * 16 + hsel * 8 + r], rowsum[m][r]);
    }
    __syncthreads();
    if (tid < SROWS) att[(size_t)b * S_DIM + s0 + tid] = ldsAtt[tid];
}

// ---- Kernel 4: softmax over S per batch ----------------------------------
__global__ __launch_bounds__(256) void softmax_kernel(const float* __restrict__ att,
                                                      float* __restrict__ alpha) {
    __shared__ float red[8];
    const int b = blockIdx.x, tid = threadIdx.x;
    const float* a = att + (size_t)b * S_DIM;
    float v0 = a[tid], v1 = a[tid + 256], v2 = a[tid + 512], v3 = a[tid + 768];

    float mx = fmaxf(fmaxf(v0, v1), fmaxf(v2, v3));
    #pragma unroll
    for (int off = 16; off; off >>= 1) mx = fmaxf(mx, __shfl_xor(mx, off, 32));
    if ((tid & 31) == 0) red[tid >> 5] = mx;
    __syncthreads();
    float m = red[0];
    #pragma unroll
    for (int i = 1; i < 8; ++i) m = fmaxf(m, red[i]);
    __syncthreads();

    float e0 = expf(v0 - m), e1 = expf(v1 - m), e2 = expf(v2 - m), e3 = expf(v3 - m);
    float s = e0 + e1 + e2 + e3;
    #pragma unroll
    for (int off = 16; off; off >>= 1) s += __shfl_xor(s, off, 32);
    if ((tid & 31) == 0) red[tid >> 5] = s;
    __syncthreads();
    float tot = red[0];
    #pragma unroll
    for (int i = 1; i < 8; ++i) tot += red[i];
    const float inv = 1.0f / tot;

    float* o = alpha + (size_t)b * S_DIM;
    o[tid] = e0 * inv; o[tid + 256] = e1 * inv; o[tid + 512] = e2 * inv; o[tid + 768] = e3 * inv;
}

// ---- Kernel 5: context[b,h] = sum_s alpha[b,s] * enc[b,s,h] --------------
__global__ __launch_bounds__(256) void context_kernel(const float* __restrict__ enc,
                                                      const float* __restrict__ alpha,
                                                      float* __restrict__ ctx) {
    const int b = blockIdx.y;
    const int h = blockIdx.x * 256 + threadIdx.x;
    const float* e  = enc + (size_t)b * S_DIM * H_DIM + h;
    const float* al = alpha + (size_t)b * S_DIM;
    float acc = 0.f;
    #pragma unroll 4
    for (int s = 0; s < S_DIM; ++s)
        acc = fmaf(al[s], e[(size_t)s * H_DIM], acc);
    ctx[b * H_DIM + h] = acc;
}

// ---------------------------------------------------------------------------
extern "C" void kernel_launch(void* const* d_in, const int* in_sizes, int n_in,
                              void* d_out, int out_size, void* d_ws, size_t ws_size,
                              hipStream_t stream) {
    const float* dh  = (const float*)d_in[0];   // [B,H]
    const float* enc = (const float*)d_in[1];   // [B,S,H]
    const float* Uw  = (const float*)d_in[2];   // [H,H]
    const float* Ww  = (const float*)d_in[3];   // [H,H]
    const float* vw  = (const float*)d_in[4];   // [H]

    float* ctx   = (float*)d_out;                       // [B,H]
    float* alpha = (float*)d_out + B_DIM * H_DIM;       // [B,S]

    uint8_t*  ws  = (uint8_t*)d_ws;
    uint32_t* Wbf = (uint32_t*)ws;                                          // 2 MB
    float*    Uh  = (float*)(ws + (size_t)H_DIM * H_DIM * 2);               // 128 KB
    float*    att = (float*)(ws + (size_t)H_DIM * H_DIM * 2 + B_DIM * H_DIM * 4);

    convw_kernel<<<(H_DIM * H_DIM / 4) / 256, 256, 0, stream>>>(Ww, Wbf);
    uh_kernel<<<dim3(H_DIM / 256, B_DIM), 256, 0, stream>>>(dh, Uw, Uh);

    size_t shB = (size_t)(SROWS * PADU) * 4 + SROWS * 4;   // 132,352 bytes LDS
    attn_scores_kernel<<<dim3(S_DIM / SROWS, B_DIM), 512, shB, stream>>>(
        enc, Wbf, Uh, vw, att);

    softmax_kernel<<<B_DIM, 256, 0, stream>>>(att, alpha);
    context_kernel<<<dim3(H_DIM / 256, B_DIM), 256, 0, stream>>>(enc, alpha, ctx);
}
